// GraphSAGE_15101105013216
// MI455X (gfx1250) — matrix operand order repaired and checked by reading, not compiled, not run on hardware
//
#include <hip/hip_runtime.h>

// ---------------------------------------------------------------------------
// GraphSAGE hetero forward for gfx1250 (MI455X).
//  - All GEMMs: v_wmma_f32_16x16x32_bf16 (f32 accumulate).
//  - Weights pre-converted to bf16 + transposed once per call -> B fragment is
//    two contiguous b128 loads per lane per K-chunk (no per-iter conversion).
//  - Intermediate features stored bf16 (halves edge-gather HBM traffic);
//    aggregation sums stay f32 via global_atomic_add_f32.
//  - bf16 A tiles staged into LDS with global_load_async_to_lds_b32 +
//    s_wait_asynccnt (CDNA5 async path).
// ---------------------------------------------------------------------------

#define HD 128  // hidden dim == Hout of every WMMA GEMM

typedef __attribute__((ext_vector_type(16))) __bf16    v16bf;
typedef __attribute__((ext_vector_type(8)))  unsigned  v8u;
typedef __attribute__((ext_vector_type(8)))  float     v8f;

union Frag16 { v8u u32; v16bf b; };

__device__ __forceinline__ unsigned short f2bfu(float f) {
    union { float f; unsigned u; } v; v.f = f;
    return (unsigned short)((v.u + 0x7FFFu + ((v.u >> 16) & 1u)) >> 16);  // RNE
}
__device__ __forceinline__ float bf2f(unsigned short u) {
    union { unsigned u; float f; } v; v.u = ((unsigned)u) << 16;
    return v.f;
}

// Y(bf16)[N,128] = op( A[rows,K] @ Wt^T (+bias) (+beta*Y) (relu) )
// Wt is bf16, transposed: Wt[col*K + k].  A is f32 (ABF16=0) or bf16 (ABF16=1).
// Block = 256 thr = 8 waves; wave w owns output cols [16w,16w+16); 16-row stripe.
template <int ABF16>
__global__ __launch_bounds__(256)
void wmma_gemm128(const void* __restrict__ A_, const int* __restrict__ rowIdx,
                  const unsigned short* __restrict__ Wt, const float* __restrict__ bias,
                  unsigned short* __restrict__ Y, int K, int beta, int relu)
{
    // 16 rows x 32 bf16 = 16 dwords/row, padded to 20 (80B) -> b128 reads are
    // 16B-aligned and the 8 lane-group fragment reads tile all 64 banks.
    __shared__ __attribute__((aligned(16))) unsigned As32[16][20];
    __shared__ int rows[16];

    const int t       = threadIdx.x;
    const int lane    = t & 31;
    const int wave    = t >> 5;
    const int rowBase = blockIdx.x * 16;

    if (t < 16) {
        int gr = rowBase + t;
        rows[t] = rowIdx ? rowIdx[gr] : gr;
    }
    __syncthreads();

    const int col  = wave * 16 + (lane & 15);
    const int kbA  = (lane >> 4) * 16;   // A-fragment K sub-base (contiguous, matches B)
    const int kbB  = (lane >> 4) * 16;   // B-fragment K sub-base (contiguous, matches A)
    const int arow = lane & 15;
    const int r    = t >> 4;             // tile row this thread stages
    const int p    = t & 15;             // dword (bf16 pair) within the row

    v8f acc = {};

    for (int kc = 0; kc < K; kc += 32) {
        // ---- stage A tile (16x32 bf16 = 256 dwords, 1 dword/thread) ----
        if (ABF16) {
            const unsigned short* Abf = (const unsigned short*)A_;
            const unsigned* gsrc =
                (const unsigned*)(Abf + (long)rows[r] * K + kc) + p;
            unsigned ldsoff = (unsigned)(unsigned long long)&As32[r][p];
            asm volatile("global_load_async_to_lds_b32 %0, %1, off"
                         :: "v"(ldsoff), "v"(gsrc) : "memory");
            asm volatile("s_wait_asynccnt 0x0" ::: "memory");
        } else {
            const float* Af = (const float*)A_;
            const float2* ar = (const float2*)(Af + (long)rows[r] * K + kc);
            float2 v = ar[p];
            As32[r][p] = (unsigned)f2bfu(v.x) | ((unsigned)f2bfu(v.y) << 16);
        }
        if (kc + 32 < K)
            __builtin_prefetch(Wt + (long)col * K + kc + 32 + kbB, 0, 3);
        __syncthreads();

        // ---- fragments ----
        Frag16 a, b;
        {
            const int d0 = kbA >> 1;                       // 0 or 8
#pragma unroll
            for (int j = 0; j < 4; ++j) {
                a.u32[j]     = As32[arow][d0 + j];         // K = kbA..kbA+7
                a.u32[4 + j] = As32[arow][d0 + 4 + j];     // K = kbA+8..kbA+15
            }
            const unsigned* wp =
                (const unsigned*)(Wt + (long)col * K + kc + kbB);
#pragma unroll
            for (int j = 0; j < 8; ++j) b.u32[j] = wp[j];  // K = kbB..kbB+15
        }
        acc = __builtin_amdgcn_wmma_f32_16x16x32_bf16(
                  false, a.b, false, b.b, (short)0, acc, false, false);
        __syncthreads();
    }

    const float bv = bias ? bias[col] : 0.0f;
#pragma unroll
    for (int rr = 0; rr < 8; ++rr) {
        const long o = (long)(rowBase + (lane >> 4) * 8 + rr) * HD + col;
        float v = acc[rr] + bv;
        if (beta) v += bf2f(Y[o]);
        if (relu) v = fmaxf(v, 0.0f);
        Y[o] = f2bfu(v);
    }
}

// Wt[c*K + k] = bf16(W[k*128 + c])  (weights are [K,128] row-major)
__global__ __launch_bounds__(256)
void weight_t_bf16(const float* __restrict__ W, unsigned short* __restrict__ Wt, int K)
{
    const long i = (long)blockIdx.x * 256 + threadIdx.x;
    if (i >= (long)K * HD) return;
    const int k = (int)(i >> 7), c = (int)(i & 127);
    Wt[(long)c * K + k] = f2bfu(W[i]);
}

// One wave per edge; lane reads 4 bf16 feats (8B) and does 4 f32 atomics.
__global__ __launch_bounds__(256)
void edge_scatter(const unsigned short* __restrict__ X, const int* __restrict__ src,
                  const int* __restrict__ dst, float* __restrict__ sum,
                  float* __restrict__ cnt, int E)
{
    const long gid  = (long)blockIdx.x * 256 + threadIdx.x;
    const int  e    = (int)(gid >> 5);
    const int  lane = (int)(gid & 31);
    if (e >= E) return;
    const int s = src[e], d = dst[e];
    const unsigned* xr = (const unsigned*)(X + (long)s * HD + lane * 4);
    float*          sr = sum + (long)d * HD + lane * 4;
    const unsigned u0 = xr[0], u1 = xr[1];
    atomicAdd(sr + 0, bf2f((unsigned short)(u0 & 0xffff)));
    atomicAdd(sr + 1, bf2f((unsigned short)(u0 >> 16)));
    atomicAdd(sr + 2, bf2f((unsigned short)(u1 & 0xffff)));
    atomicAdd(sr + 3, bf2f((unsigned short)(u1 >> 16)));
    if (lane == 0) atomicAdd(cnt + d, 1.0f);
}

__global__ __launch_bounds__(256)
void finalize_mean_bf(const float* __restrict__ sum, const float* __restrict__ cnt,
                      unsigned short* __restrict__ ob, long n)
{
    const long i = (long)blockIdx.x * 256 + threadIdx.x;
    if (i >= n) return;
    ob[i] = f2bfu(sum[i] * (1.0f / fmaxf(cnt[i >> 7], 1.0f)));
}

__global__ __launch_bounds__(256)
void zero_f32(float* __restrict__ pz, long n)
{
    const long i = (long)blockIdx.x * 256 + threadIdx.x;
    if (i < n) pz[i] = 0.0f;
}

// out[b] = dot(h[b,:128](bf16), W2) + b2 ; one wave32 per row.
__global__ __launch_bounds__(256)
void head_gemv(const unsigned short* __restrict__ Hm, const float* __restrict__ W2,
               const float* __restrict__ b2, float* __restrict__ out, int B)
{
    const int lane = threadIdx.x & 31;
    const int row  = blockIdx.x * 8 + (threadIdx.x >> 5);
    if (row >= B) return;
    const unsigned* hr = (const unsigned*)(Hm + (long)row * HD + lane * 4);
    const float*    wr = W2 + lane * 4;
    const unsigned u0 = hr[0], u1 = hr[1];
    float s = bf2f((unsigned short)(u0 & 0xffff)) * wr[0]
            + bf2f((unsigned short)(u0 >> 16))    * wr[1]
            + bf2f((unsigned short)(u1 & 0xffff)) * wr[2]
            + bf2f((unsigned short)(u1 >> 16))    * wr[3];
#pragma unroll
    for (int off = 16; off > 0; off >>= 1)
        s += __shfl_down(s, off, 32);
    if (lane == 0) out[row] = s + b2[0];
}

extern "C" void kernel_launch(void* const* d_in, const int* in_sizes, int n_in,
                              void* d_out, int out_size, void* d_ws, size_t ws_size,
                              hipStream_t stream)
{
    (void)in_sizes; (void)n_in; (void)out_size; (void)ws_size;

    const float* x_drug      = (const float*)d_in[0];
    const float* x_protein   = (const float*)d_in[1];
    const int*   edge_dp_src = (const int*)  d_in[2];
    const int*   edge_dp_dst = (const int*)  d_in[3];
    const int*   edge_pd_src = (const int*)  d_in[4];
    const int*   edge_pd_dst = (const int*)  d_in[5];
    const int*   drug_idx    = (const int*)  d_in[6];
    const int*   protein_idx = (const int*)  d_in[7];
    const float* W_dlin      = (const float*)d_in[8];
    const float* b_dlin      = (const float*)d_in[9];
    const float* W_plin      = (const float*)d_in[10];
    const float* b_plin      = (const float*)d_in[11];
    const float* W_dp_l      = (const float*)d_in[12];
    const float* b_dp_l      = (const float*)d_in[13];
    const float* W_dp_r      = (const float*)d_in[14];
    const float* W_pd_l      = (const float*)d_in[15];
    const float* b_pd_l      = (const float*)d_in[16];
    const float* W_pd_r      = (const float*)d_in[17];
    const float* W1          = (const float*)d_in[18];
    const float* b1          = (const float*)d_in[19];
    const float* W2          = (const float*)d_in[20];
    const float* b2          = (const float*)d_in[21];
    float*       out         = (float*)d_out;

    const int ND = 100000, NP = 50000, DD = 128, DP = 1024, E = 1000000, Bn = 65536;
    typedef unsigned short u16;

    // ---- workspace partition (256B-aligned carve) ----
    char* wp_ = (char*)d_ws;
    auto carve = [&](size_t bytes) -> char* {
        char* r = wp_; wp_ += (bytes + 255) & ~(size_t)255; return r;
    };
    u16*   xd_bf   = (u16*)  carve((size_t)ND * HD * 2);
    u16*   xp_bf   = (u16*)  carve((size_t)NP * HD * 2);
    u16*   xdo_bf  = (u16*)  carve((size_t)ND * HD * 2);
    u16*   xpo_bf  = (u16*)  carve((size_t)NP * HD * 2);
    u16*   aggp_bf = (u16*)  carve((size_t)NP * HD * 2);
    u16*   aggd_bf = (u16*)  carve((size_t)ND * HD * 2);
    float* sums    = (float*)carve(((size_t)NP * HD + (size_t)ND * HD + NP + ND) * 4);
    float* sum_p   = sums;
    float* sum_d   = sum_p + (long)NP * HD;
    float* cnt_p   = sum_d + (long)ND * HD;
    float* cnt_d   = cnt_p + NP;
    u16*   h_bf    = (u16*)sum_p;  // alias: sums are dead before h is written
    u16*   Wt_dlin = (u16*)carve((size_t)DD * HD * 2);
    u16*   Wt_plin = (u16*)carve((size_t)DP * HD * 2);
    u16*   Wt_dpl  = (u16*)carve((size_t)HD * HD * 2);
    u16*   Wt_dpr  = (u16*)carve((size_t)HD * HD * 2);
    u16*   Wt_pdl  = (u16*)carve((size_t)HD * HD * 2);
    u16*   Wt_pdr  = (u16*)carve((size_t)HD * HD * 2);
    u16*   Wt_1a   = (u16*)carve((size_t)HD * HD * 2);
    u16*   Wt_1b   = (u16*)carve((size_t)HD * HD * 2);

    // 1. zero aggregation sums + counts
    {
        const long zn = (long)NP * HD + (long)ND * HD + NP + ND;
        zero_f32<<<(int)((zn + 255) / 256), 256, 0, stream>>>(sums, zn);
    }

    // 2. pre-convert weights to bf16, transposed (Wt[col][k])
    const int wB = 256;
    weight_t_bf16<<<(DD * HD + wB - 1) / wB, wB, 0, stream>>>(W_dlin, Wt_dlin, DD);
    weight_t_bf16<<<(DP * HD + wB - 1) / wB, wB, 0, stream>>>(W_plin, Wt_plin, DP);
    weight_t_bf16<<<(HD * HD + wB - 1) / wB, wB, 0, stream>>>(W_dp_l, Wt_dpl, HD);
    weight_t_bf16<<<(HD * HD + wB - 1) / wB, wB, 0, stream>>>(W_dp_r, Wt_dpr, HD);
    weight_t_bf16<<<(HD * HD + wB - 1) / wB, wB, 0, stream>>>(W_pd_l, Wt_pdl, HD);
    weight_t_bf16<<<(HD * HD + wB - 1) / wB, wB, 0, stream>>>(W_pd_r, Wt_pdr, HD);
    weight_t_bf16<<<(HD * HD + wB - 1) / wB, wB, 0, stream>>>(W1,            Wt_1a, HD);
    weight_t_bf16<<<(HD * HD + wB - 1) / wB, wB, 0, stream>>>(W1 + HD * HD,  Wt_1b, HD);

    // 3. input projections (A = f32, out bf16)
    wmma_gemm128<0><<<ND / 16, 256, 0, stream>>>(x_drug,    nullptr, Wt_dlin, b_dlin, xd_bf, DD, 0, 0);
    wmma_gemm128<0><<<NP / 16, 256, 0, stream>>>(x_protein, nullptr, Wt_plin, b_plin, xp_bf, DP, 0, 0);

    // 4. edge gather/scatter (bf16 gathers, f32 atomic sums + counts)
    {
        const long st = (long)E * 32;
        const int  gb = (int)((st + 255) / 256);
        edge_scatter<<<gb, 256, 0, stream>>>(xd_bf, edge_dp_src, edge_dp_dst, sum_p, cnt_p, E);
        edge_scatter<<<gb, 256, 0, stream>>>(xp_bf, edge_pd_src, edge_pd_dst, sum_d, cnt_d, E);
    }

    // 5. mean = sum / max(cnt,1), stored bf16
    finalize_mean_bf<<<(int)(((long)NP * HD + 255) / 256), 256, 0, stream>>>(sum_p, cnt_p, aggp_bf, (long)NP * HD);
    finalize_mean_bf<<<(int)(((long)ND * HD + 255) / 256), 256, 0, stream>>>(sum_d, cnt_d, aggd_bf, (long)ND * HD);

    // 6. SAGEConv: lin_l(mean) + bias + lin_r(root)   (A = bf16, async LDS path)
    wmma_gemm128<1><<<NP / 16, 256, 0, stream>>>(aggp_bf, nullptr, Wt_dpl, b_dp_l, xpo_bf, HD, 0, 0);
    wmma_gemm128<1><<<NP / 16, 256, 0, stream>>>(xp_bf,   nullptr, Wt_dpr, nullptr, xpo_bf, HD, 1, 0);
    wmma_gemm128<1><<<ND / 16, 256, 0, stream>>>(aggd_bf, nullptr, Wt_pdl, b_pd_l, xdo_bf, HD, 0, 0);
    wmma_gemm128<1><<<ND / 16, 256, 0, stream>>>(xd_bf,   nullptr, Wt_pdr, nullptr, xdo_bf, HD, 1, 0);

    // 7. pair readout + MLP layer 1 without materializing the concat:
    //    h = relu( xdo[drug_idx] @ W1[:128] + xpo[protein_idx] @ W1[128:] + b1 )
    wmma_gemm128<1><<<Bn / 16, 256, 0, stream>>>(xdo_bf, drug_idx,    Wt_1a, b1,      h_bf, HD, 0, 0);
    wmma_gemm128<1><<<Bn / 16, 256, 0, stream>>>(xpo_bf, protein_idx, Wt_1b, nullptr, h_bf, HD, 1, 1);

    // 8. scalar head
    head_gemv<<<Bn / 8, 256, 0, stream>>>(h_bf, W2, b2, out, Bn);
}
